// DecoderRNN_25271587569677
// MI455X (gfx1250) — matrix operand order repaired and checked by reading, not compiled, not run on hardware
//
#include <hip/hip_runtime.h>

// ---------------- constants (match reference) ----------------
#define Bc 128
#define Tc 50
#define Hc 1024
#define Lc 2
#define Vc 8192
#define BT (Bc * Tc)          // 6400 rows
#define H3 (3 * Hc)           // 3072

// ---------------- WMMA types & helpers ----------------
typedef __bf16 bf16_t;
typedef bf16_t v16bf __attribute__((ext_vector_type(16)));
typedef float  v8f   __attribute__((ext_vector_type(8)));

union FragBF { v16bf v; uint4 q[2]; };

__device__ __forceinline__ unsigned short f2bf(float x) {
    unsigned int u = __float_as_uint(x);
    unsigned int r = u + 0x7FFFu + ((u >> 16) & 1u);   // round-to-nearest-even
    return (unsigned short)(r >> 16);
}

// A fragment 16x32 bf16 (M x K), row-major source, `row` already points at
// row (mbase + lane&15). Lanes 0-15: K = k..k+7 and k+16..k+23.
// Lanes 16-31: K = k+8..k+15 and k+24..k+31.   (ISA 7.12.2, 16-bit A)
__device__ __forceinline__ v16bf load_frag_a(const unsigned short* row, int lane, int k) {
    int kh = (lane >> 4) & 1;
    const uint4* p = (const uint4*)(row + k + kh * 8);
    FragBF f;
    f.q[0] = p[0];   // K k+kh*8    .. +7
    f.q[1] = p[2];   // K k+16+kh*8 .. +7
    return f.v;
}

// B fragment 32x16 bf16 (K x N) from row-major weight W[N][K]; `row` already
// points at W row (nbase + lane&15). Lanes 0-15: K = k..k+7 and k+16..k+23,
// lanes 16-31: K = k+8..k+15 and k+24..k+31.
__device__ __forceinline__ v16bf load_frag_b(const unsigned short* row, int lane, int k) {
    int kh = ((lane >> 4) & 1) << 3;
    const uint4* p = (const uint4*)(row + k + kh);
    FragBF f;
    f.q[0] = p[0];
    f.q[1] = p[2];
    return f.v;
}

__device__ __forceinline__ v8f wmma_bf16(v16bf a, v16bf b, v8f c) {
    return __builtin_amdgcn_wmma_f32_16x16x32_bf16(false, a, false, b, (short)0, c, false, false);
}

__device__ __forceinline__ float fsigmoid(float x) { return 1.f / (1.f + __expf(-x)); }

// ---------------- elementwise prep kernels ----------------
__global__ void f32_to_bf16_kernel(const float* __restrict__ in,
                                   unsigned short* __restrict__ outp, int n) {
    int i = blockIdx.x * blockDim.x + threadIdx.x;
    if (i < n) outp[i] = f2bf(in[i]);
}

// tokens[b,t] = (t==0 ? SOW=0 : target[b,t-1]); emb = relu(embedding[token])
__global__ void embed_kernel(const int* __restrict__ target,
                             const float* __restrict__ embedding,
                             unsigned short* __restrict__ embf) {
    int i = blockIdx.x * blockDim.x + threadIdx.x;      // over B*T*H
    if (i >= BT * Hc) return;
    int h  = i & (Hc - 1);
    int bt = i >> 10;                                   // /H
    int t  = bt % Tc;
    int b  = bt / Tc;
    int tok = (t == 0) ? 0 : target[b * Tc + t - 1];
    float v = embedding[(size_t)tok * Hc + h];
    embf[i] = f2bf(v > 0.f ? v : 0.f);
}

// h state init from encoder_hidden [L,B,H]; bf16 copy goes to ping buffer 0
__global__ void init_state_kernel(const float* __restrict__ enc,
                                  float* __restrict__ hf,
                                  unsigned short* __restrict__ hbf) {
    int i = blockIdx.x * blockDim.x + threadIdx.x;      // over L*B*H
    if (i >= Lc * Bc * Hc) return;
    float v = enc[i];
    hf[i] = v;
    int l = i / (Bc * Hc);
    int r = i % (Bc * Hc);
    hbf[((size_t)l * 2) * (Bc * Hc) + r] = f2bf(v);     // buffer 0 of layer l
}

__global__ void copy_hfinal_kernel(const float* __restrict__ hf, float* __restrict__ dst) {
    int i = blockIdx.x * blockDim.x + threadIdx.x;      // over L*B*H
    if (i < Lc * Bc * Hc) dst[i] = hf[i];
}

// ---------------- register-blocked, double-buffered WMMA GEMM ----------------
// C[M,N] = A[M,K] * W[N,K]^T + bias.  Each wave computes a (BM*16) x (BN*16)
// block; fragments for K-chunk k+32 are loaded into a second register set
// before the WMMAs for chunk k issue, so loads stay in flight across the
// matrix burst.  K is compile-time (1024), K % 64 == 0.
#define BM 4
#define BN 2
template <int K>
__global__ void wmma_gemm_blocked_kernel(const unsigned short* __restrict__ A,
                                         const unsigned short* __restrict__ W,
                                         const float* __restrict__ bias,
                                         float* __restrict__ C,
                                         int M, int N, int ldc) {
    int lane = threadIdx.x & 31;
    int wave = threadIdx.x >> 5;
    int nblks = N / (BN * 16);
    int mblks = M / (BM * 16);
    int blk = blockIdx.x * (blockDim.x >> 5) + wave;
    if (blk >= mblks * nblks) return;                   // wave-uniform
    int mb = blk / nblks, nb = blk - mb * nblks;
    int mbase = mb * (BM * 16), nbase = nb * (BN * 16);
    int lm = lane & 15;

    const unsigned short* arow[BM];
#pragma unroll
    for (int i = 0; i < BM; ++i)
        arow[i] = A + (size_t)(mbase + i * 16 + lm) * K;
    const unsigned short* wrow[BN];
#pragma unroll
    for (int j = 0; j < BN; ++j)
        wrow[j] = W + (size_t)(nbase + j * 16 + lm) * K;

    v8f acc[BM][BN];
#pragma unroll
    for (int i = 0; i < BM; ++i)
#pragma unroll
        for (int j = 0; j < BN; ++j) acc[i][j] = (v8f){};

    v16bf a0[BM], b0[BN], a1[BM], b1[BN];
#pragma unroll
    for (int i = 0; i < BM; ++i) a0[i] = load_frag_a(arow[i], lane, 0);
#pragma unroll
    for (int j = 0; j < BN; ++j) b0[j] = load_frag_b(wrow[j], lane, 0);

    int k = 0;
    for (; k + 64 < K; k += 64) {
#pragma unroll
        for (int i = 0; i < BM; ++i) a1[i] = load_frag_a(arow[i], lane, k + 32);
#pragma unroll
        for (int j = 0; j < BN; ++j) b1[j] = load_frag_b(wrow[j], lane, k + 32);
#pragma unroll
        for (int i = 0; i < BM; ++i)
#pragma unroll
            for (int j = 0; j < BN; ++j) acc[i][j] = wmma_bf16(a0[i], b0[j], acc[i][j]);
#pragma unroll
        for (int i = 0; i < BM; ++i) a0[i] = load_frag_a(arow[i], lane, k + 64);
#pragma unroll
        for (int j = 0; j < BN; ++j) b0[j] = load_frag_b(wrow[j], lane, k + 64);
#pragma unroll
        for (int i = 0; i < BM; ++i)
#pragma unroll
            for (int j = 0; j < BN; ++j) acc[i][j] = wmma_bf16(a1[i], b1[j], acc[i][j]);
    }
    // tail: chunks k (in set 0) and k+32
#pragma unroll
    for (int i = 0; i < BM; ++i) a1[i] = load_frag_a(arow[i], lane, k + 32);
#pragma unroll
    for (int j = 0; j < BN; ++j) b1[j] = load_frag_b(wrow[j], lane, k + 32);
#pragma unroll
    for (int i = 0; i < BM; ++i)
#pragma unroll
        for (int j = 0; j < BN; ++j) acc[i][j] = wmma_bf16(a0[i], b0[j], acc[i][j]);
#pragma unroll
    for (int i = 0; i < BM; ++i)
#pragma unroll
        for (int j = 0; j < BN; ++j) acc[i][j] = wmma_bf16(a1[i], b1[j], acc[i][j]);

    int moff = (lane >> 4) << 3;
#pragma unroll
    for (int j = 0; j < BN; ++j) {
        int n = nbase + j * 16 + lm;
        float bn = bias ? bias[n] : 0.f;
#pragma unroll
        for (int i = 0; i < BM; ++i) {
            float* crow = C + (size_t)(mbase + i * 16 + moff) * ldc + n;
#pragma unroll
            for (int e = 0; e < 8; ++e) crow[(size_t)e * ldc] = acc[i][j][e] + bn;
        }
    }
}

// ---------------- GRU layer 0 step: gh GEMM + gates (gi precomputed) ----------------
// 512 tiles: mtiles=8 (B/16), ntiles=64 (H/16). gi0 already contains b_ih[0].
// Double-buffered fragment loads to shorten the sequential critical path.
__global__ void gru_step_l0_kernel(const unsigned short* __restrict__ h_src, // [B,H] bf16
                                   unsigned short* __restrict__ h_dst,       // [B,H] bf16
                                   const unsigned short* __restrict__ whh,   // [3H,H] bf16
                                   const float* __restrict__ gi0,            // [B*T,3H]
                                   const float* __restrict__ bhh,            // [3H]
                                   float* __restrict__ hf,                   // [B,H] f32
                                   int t) {
    int lane = threadIdx.x & 31;
    int wave = threadIdx.x >> 5;
    int tile = blockIdx.x * (blockDim.x >> 5) + wave;   // 0..511
    const int ntiles = Hc >> 4;                         // 64
    int mt = tile / ntiles, nt = tile - mt * ntiles;
    int mbase = mt << 4, nbase = nt << 4;
    int lm = lane & 15;
    int n = nbase + lm;
    const unsigned short* arow = h_src + (size_t)(mbase + lm) * Hc;
    const unsigned short* wr = whh + (size_t)n * Hc;
    const unsigned short* wz = whh + (size_t)(n + Hc) * Hc;
    const unsigned short* wn = whh + (size_t)(n + 2 * Hc) * Hc;
    v8f ar = {}, az = {}, an = {};

    v16bf a0, r0, z0, n0, a1, r1, z1, n1;
    a0 = load_frag_a(arow, lane, 0);
    r0 = load_frag_b(wr, lane, 0);
    z0 = load_frag_b(wz, lane, 0);
    n0 = load_frag_b(wn, lane, 0);
    int k = 0;
    for (; k + 64 < Hc; k += 64) {
        a1 = load_frag_a(arow, lane, k + 32);
        r1 = load_frag_b(wr, lane, k + 32);
        z1 = load_frag_b(wz, lane, k + 32);
        n1 = load_frag_b(wn, lane, k + 32);
        ar = wmma_bf16(a0, r0, ar);
        az = wmma_bf16(a0, z0, az);
        an = wmma_bf16(a0, n0, an);
        a0 = load_frag_a(arow, lane, k + 64);
        r0 = load_frag_b(wr, lane, k + 64);
        z0 = load_frag_b(wz, lane, k + 64);
        n0 = load_frag_b(wn, lane, k + 64);
        ar = wmma_bf16(a1, r1, ar);
        az = wmma_bf16(a1, z1, az);
        an = wmma_bf16(a1, n1, an);
    }
    a1 = load_frag_a(arow, lane, k + 32);
    r1 = load_frag_b(wr, lane, k + 32);
    z1 = load_frag_b(wz, lane, k + 32);
    n1 = load_frag_b(wn, lane, k + 32);
    ar = wmma_bf16(a0, r0, ar);
    az = wmma_bf16(a0, z0, az);
    an = wmma_bf16(a0, n0, an);
    ar = wmma_bf16(a1, r1, ar);
    az = wmma_bf16(a1, z1, az);
    an = wmma_bf16(a1, n1, an);

    float bhr = bhh[n], bhz = bhh[n + Hc], bhn = bhh[n + 2 * Hc];
    int mrow = mbase + ((lane >> 4) << 3);
#pragma unroll
    for (int e = 0; e < 8; ++e) {
        int b = mrow + e;
        const float* gi = gi0 + ((size_t)b * Tc + t) * H3;
        float r = fsigmoid(gi[n] + ar[e] + bhr);
        float z = fsigmoid(gi[n + Hc] + az[e] + bhz);
        float nn = tanhf(gi[n + 2 * Hc] + r * (an[e] + bhn));
        size_t idx = (size_t)b * Hc + n;
        float hp = hf[idx];
        float hnew = (1.f - z) * nn + z * hp;
        hf[idx] = hnew;
        h_dst[idx] = f2bf(hnew);
    }
}

// ---------------- GRU layer 1 step: gi + gh GEMMs + gates, emit top output ----------------
__global__ void gru_step_l1_kernel(const unsigned short* __restrict__ x_src,  // new h0 [B,H]
                                   const unsigned short* __restrict__ h_src,  // h1 [B,H]
                                   unsigned short* __restrict__ h_dst,        // h1 out [B,H]
                                   const unsigned short* __restrict__ wih,    // [3H,H]
                                   const unsigned short* __restrict__ whh,    // [3H,H]
                                   const float* __restrict__ bih,
                                   const float* __restrict__ bhh,
                                   float* __restrict__ hf,                    // h1 f32
                                   unsigned short* __restrict__ outs,         // [B*T,H] bf16
                                   int t) {
    int lane = threadIdx.x & 31;
    int wave = threadIdx.x >> 5;
    int tile = blockIdx.x * (blockDim.x >> 5) + wave;   // 0..511
    const int ntiles = Hc >> 4;
    int mt = tile / ntiles, nt = tile - mt * ntiles;
    int mbase = mt << 4, nbase = nt << 4;
    int lm = lane & 15;
    int n = nbase + lm;
    const unsigned short* xrow = x_src + (size_t)(mbase + lm) * Hc;
    const unsigned short* hrow = h_src + (size_t)(mbase + lm) * Hc;
    const unsigned short* wrow[6];
    wrow[0] = wih + (size_t)n * Hc;
    wrow[1] = wih + (size_t)(n + Hc) * Hc;
    wrow[2] = wih + (size_t)(n + 2 * Hc) * Hc;
    wrow[3] = whh + (size_t)n * Hc;
    wrow[4] = whh + (size_t)(n + Hc) * Hc;
    wrow[5] = whh + (size_t)(n + 2 * Hc) * Hc;

    v8f acc[6];
#pragma unroll
    for (int g = 0; g < 6; ++g) acc[g] = (v8f){};

    v16bf ax0, ah0, ax1, ah1, w0[6], w1[6];
    ax0 = load_frag_a(xrow, lane, 0);
    ah0 = load_frag_a(hrow, lane, 0);
#pragma unroll
    for (int g = 0; g < 6; ++g) w0[g] = load_frag_b(wrow[g], lane, 0);
    int k = 0;
    for (; k + 64 < Hc; k += 64) {
        ax1 = load_frag_a(xrow, lane, k + 32);
        ah1 = load_frag_a(hrow, lane, k + 32);
#pragma unroll
        for (int g = 0; g < 6; ++g) w1[g] = load_frag_b(wrow[g], lane, k + 32);
#pragma unroll
        for (int g = 0; g < 3; ++g) acc[g] = wmma_bf16(ax0, w0[g], acc[g]);
#pragma unroll
        for (int g = 3; g < 6; ++g) acc[g] = wmma_bf16(ah0, w0[g], acc[g]);
        ax0 = load_frag_a(xrow, lane, k + 64);
        ah0 = load_frag_a(hrow, lane, k + 64);
#pragma unroll
        for (int g = 0; g < 6; ++g) w0[g] = load_frag_b(wrow[g], lane, k + 64);
#pragma unroll
        for (int g = 0; g < 3; ++g) acc[g] = wmma_bf16(ax1, w1[g], acc[g]);
#pragma unroll
        for (int g = 3; g < 6; ++g) acc[g] = wmma_bf16(ah1, w1[g], acc[g]);
    }
    ax1 = load_frag_a(xrow, lane, k + 32);
    ah1 = load_frag_a(hrow, lane, k + 32);
#pragma unroll
    for (int g = 0; g < 6; ++g) w1[g] = load_frag_b(wrow[g], lane, k + 32);
#pragma unroll
    for (int g = 0; g < 3; ++g) acc[g] = wmma_bf16(ax0, w0[g], acc[g]);
#pragma unroll
    for (int g = 3; g < 6; ++g) acc[g] = wmma_bf16(ah0, w0[g], acc[g]);
#pragma unroll
    for (int g = 0; g < 3; ++g) acc[g] = wmma_bf16(ax1, w1[g], acc[g]);
#pragma unroll
    for (int g = 3; g < 6; ++g) acc[g] = wmma_bf16(ah1, w1[g], acc[g]);

    float bir = bih[n], biz = bih[n + Hc], bin_ = bih[n + 2 * Hc];
    float bhr = bhh[n], bhz = bhh[n + Hc], bhn = bhh[n + 2 * Hc];
    int mrow = mbase + ((lane >> 4) << 3);
#pragma unroll
    for (int e = 0; e < 8; ++e) {
        int b = mrow + e;
        float r = fsigmoid((acc[0][e] + bir) + (acc[3][e] + bhr));
        float z = fsigmoid((acc[1][e] + biz) + (acc[4][e] + bhz));
        float nn = tanhf((acc[2][e] + bin_) + r * (acc[5][e] + bhn));
        size_t idx = (size_t)b * Hc + n;
        float hp = hf[idx];
        float hnew = (1.f - z) * nn + z * hp;
        hf[idx] = hnew;
        h_dst[idx] = f2bf(hnew);
        outs[((size_t)b * Tc + t) * Hc + n] = f2bf(hnew);
    }
}

// ---------------- row log-softmax over V=8192, in place ----------------
__global__ void log_softmax_kernel(float* __restrict__ x) {
    __shared__ float red[256];
    int row = blockIdx.x;
    float* p = x + (size_t)row * Vc;
    float m = -INFINITY;
    for (int i = threadIdx.x; i < Vc; i += 256) m = fmaxf(m, p[i]);
    red[threadIdx.x] = m;
    __syncthreads();
    for (int s = 128; s > 0; s >>= 1) {
        if (threadIdx.x < s) red[threadIdx.x] = fmaxf(red[threadIdx.x], red[threadIdx.x + s]);
        __syncthreads();
    }
    m = red[0];
    __syncthreads();
    float sum = 0.f;
    for (int i = threadIdx.x; i < Vc; i += 256) sum += __expf(p[i] - m);
    red[threadIdx.x] = sum;
    __syncthreads();
    for (int s = 128; s > 0; s >>= 1) {
        if (threadIdx.x < s) red[threadIdx.x] += red[threadIdx.x + s];
        __syncthreads();
    }
    float lse = m + __logf(red[0]);
    for (int i = threadIdx.x; i < Vc; i += 256) p[i] = p[i] - lse;
}

// ---------------- host launcher ----------------
extern "C" void kernel_launch(void* const* d_in, const int* in_sizes, int n_in,
                              void* d_out, int out_size, void* d_ws, size_t ws_size,
                              hipStream_t stream) {
    (void)in_sizes; (void)n_in; (void)out_size; (void)ws_size;
    // inputs (setup_inputs order)
    const float* enc_hidden = (const float*)d_in[1];   // [L,B,H]
    const int*   target     = (const int*)d_in[2];     // [B,T]
    const float* embedding  = (const float*)d_in[3];   // [V,H]
    const float* w_ih       = (const float*)d_in[4];   // [L,3H,H]
    const float* w_hh       = (const float*)d_in[5];   // [L,3H,H]
    const float* b_ih       = (const float*)d_in[6];   // [L,3H]
    const float* b_hh       = (const float*)d_in[7];   // [L,3H]
    const float* out_w      = (const float*)d_in[8];   // [V,H]
    const float* out_b      = (const float*)d_in[9];   // [V]
    float* out = (float*)d_out;                        // [B,T,V] log_probs + [L,B,H] h_final

    // workspace carve-up
    size_t off = 0;
    auto alloc = [&](size_t bytes) -> char* {
        char* p = (char*)d_ws + off;
        off = (off + bytes + 255) & ~(size_t)255;
        return p;
    };
    unsigned short* wih_bf  = (unsigned short*)alloc((size_t)Lc * H3 * Hc * 2);
    unsigned short* whh_bf  = (unsigned short*)alloc((size_t)Lc * H3 * Hc * 2);
    unsigned short* outw_bf = (unsigned short*)alloc((size_t)Vc * Hc * 2);
    unsigned short* emb_bf  = (unsigned short*)alloc((size_t)BT * Hc * 2);
    unsigned short* outs_bf = (unsigned short*)alloc((size_t)BT * Hc * 2);
    float*          h_f32   = (float*)alloc((size_t)Lc * Bc * Hc * 4);
    unsigned short* h_bf    = (unsigned short*)alloc((size_t)Lc * 2 * Bc * Hc * 2); // ping-pong
    float*          gi0     = (float*)alloc((size_t)BT * H3 * 4);

    auto hb = [&](int l, int buf) { return h_bf + ((size_t)l * 2 + buf) * (Bc * Hc); };

    const int TPB = 256;
    // 1) weight / activation conversion to bf16
    {
        int n = Lc * H3 * Hc;
        f32_to_bf16_kernel<<<(n + TPB - 1) / TPB, TPB, 0, stream>>>(w_ih, wih_bf, n);
        f32_to_bf16_kernel<<<(n + TPB - 1) / TPB, TPB, 0, stream>>>(w_hh, whh_bf, n);
        int nw = Vc * Hc;
        f32_to_bf16_kernel<<<(nw + TPB - 1) / TPB, TPB, 0, stream>>>(out_w, outw_bf, nw);
    }
    // 2) gather + relu embeddings
    {
        int n = BT * Hc;
        embed_kernel<<<(n + TPB - 1) / TPB, TPB, 0, stream>>>(target, embedding, emb_bf);
    }
    // 3) init hidden state
    {
        int n = Lc * Bc * Hc;
        init_state_kernel<<<(n + TPB - 1) / TPB, TPB, 0, stream>>>(enc_hidden, h_f32, h_bf);
    }
    // 4) hoisted layer-0 input GEMM: gi0[B*T,3H] = emb @ w_ih0^T + b_ih0
    {
        int blks = (BT / (BM * 16)) * (H3 / (BN * 16));   // 100 * 96
        wmma_gemm_blocked_kernel<Hc><<<(blks + 7) / 8, TPB, 0, stream>>>(
            emb_bf, wih_bf, b_ih, gi0, BT, H3, H3);
    }
    // 5) sequential GRU scan (2 launches per step, ping-pong bf16 state)
    for (int t = 0; t < Tc; ++t) {
        int src = t & 1, dst = 1 - src;
        gru_step_l0_kernel<<<64, TPB, 0, stream>>>(
            hb(0, src), hb(0, dst), whh_bf, gi0, b_hh, h_f32, t);
        gru_step_l1_kernel<<<64, TPB, 0, stream>>>(
            hb(0, dst), hb(1, src), hb(1, dst),
            wih_bf + (size_t)H3 * Hc, whh_bf + (size_t)H3 * Hc,
            b_ih + H3, b_hh + H3,
            h_f32 + (size_t)Bc * Hc, outs_bf, t);
    }
    // 6) vocab projection: logits[B*T,V] = outs @ out_w^T + out_b  (written to d_out)
    {
        int blks = (BT / (BM * 16)) * (Vc / (BN * 16));   // 100 * 256
        wmma_gemm_blocked_kernel<Hc><<<(blks + 7) / 8, TPB, 0, stream>>>(
            outs_bf, outw_bf, out_b, out, BT, Vc, Vc);
    }
    // 7) in-place log-softmax over V per (b,t) row
    log_softmax_kernel<<<BT, TPB, 0, stream>>>(out);
    // 8) h_final
    {
        int n = Lc * Bc * Hc;
        copy_hfinal_kernel<<<(n + TPB - 1) / TPB, TPB, 0, stream>>>(
            h_f32, out + (size_t)BT * Vc);
    }
}
